// GSAAttention_62474594287823
// MI455X (gfx1250) — matrix operand order repaired and checked by reading, not compiled, not run on hardware
//
#include <hip/hip_runtime.h>
#include <hip/hip_bf16.h>
#include <math.h>

// ---------------------------------------------------------------------------
// Types for CDNA5 WMMA (wave32): v_wmma_f32_16x16x32_bf16
// ---------------------------------------------------------------------------
typedef __attribute__((ext_vector_type(16))) __bf16 v16bf;
typedef __attribute__((ext_vector_type(8)))  __bf16 v8bf;
typedef __attribute__((ext_vector_type(4)))  __bf16 v4bf;
typedef __attribute__((ext_vector_type(8)))  float  v8f;

#define DIMC    256
#define HEADS   8
#define HDIM    32
#define BATCH   2
#define SEQN    4096          // 64*64
#define SEQM    1024          // 32*32 after stride-2 conv
#define KCONV   1024          // 256 * 2 * 2 im2col depth
#define SCALE_F 0.17677669529663687f   // 32^-0.5, folded into q_w

// ---------------------------------------------------------------------------
// Async global->LDS staging (CDNA5 GLOBAL_LOAD_ASYNC_TO_LDS_B128 + ASYNCcnt).
// Param type per hipcc diagnostic: int __attribute__((vector_size(16))) * in
// global AS; LDS side symmetric in AS(3).
// ---------------------------------------------------------------------------
#if defined(__has_builtin)
#if __has_builtin(__builtin_amdgcn_global_load_async_to_lds_b128) && \
    __has_builtin(__builtin_amdgcn_s_wait_asynccnt)
#define HAVE_ASYNC_LDS 1
#endif
#endif
#ifndef HAVE_ASYNC_LDS
#define HAVE_ASYNC_LDS 0
#endif

typedef int v4i_vs __attribute__((vector_size(16)));

__device__ __forceinline__ void async_copy16(const __bf16* g, __bf16* l) {
#if HAVE_ASYNC_LDS
  __builtin_amdgcn_global_load_async_to_lds_b128(
      (__attribute__((address_space(1))) v4i_vs*)g,
      (__attribute__((address_space(3))) v4i_vs*)l, 0, 0);
#else
  *(v8bf*)l = *(const v8bf*)g;
#endif
}
__device__ __forceinline__ void async_wait_all() {
#if HAVE_ASYNC_LDS
  __builtin_amdgcn_s_wait_asynccnt(0);
#endif
}

__device__ __forceinline__ v16bf cat16(v8bf a, v8bf b) {
  v16bf r;
#pragma unroll
  for (int i = 0; i < 8; ++i) { r[i] = a[i]; r[i + 8] = b[i]; }
  return r;
}

// A-fragment (16x32 bf16, MxK). p = &A[row = lane&15][k0 + (lane>>4)*16].
// Layout: lanes 0-15 hold K {0..15}; lanes 16-31 hold K {16..31} (matches B-fragment).
__device__ __forceinline__ v16bf load_afrag(const __bf16* p) {
  return cat16(*(const v8bf*)p, *(const v8bf*)(p + 8));
}

// B-fragment (32x16 bf16, KxN): 16 contiguous bf16 at
// p = &Bcol[col = lane&15][k0 + (lane>>4)*16]  (column stored contiguously).
__device__ __forceinline__ v16bf load_bfrag(const __bf16* p) {
  return cat16(*(const v8bf*)p, *(const v8bf*)(p + 8));
}

// ---------------------------------------------------------------------------
// Elementwise conversion / packing kernels
// ---------------------------------------------------------------------------
__global__ void k_f32_to_bf16_v4(const float* __restrict__ src, __bf16* __restrict__ dst,
                                 float scale, int n4) {
  int i = blockIdx.x * blockDim.x + threadIdx.x;
  if (i >= n4) return;
  float4 v = ((const float4*)src)[i];
  v4bf o;
  o[0] = (__bf16)(v.x * scale); o[1] = (__bf16)(v.y * scale);
  o[2] = (__bf16)(v.z * scale); o[3] = (__bf16)(v.w * scale);
  ((v4bf*)dst)[i] = o;
}

// sr_w [oc][c][ky][kx] -> wsr [oc][s*256 + c], s = ky*2+kx   (bf16)
__global__ void k_pack_srw(const float* __restrict__ src, __bf16* __restrict__ dst, int n) {
  int i = blockIdx.x * blockDim.x + threadIdx.x;
  if (i >= n) return;
  int oc = i >> 10;
  int kk = i & 1023;
  int s  = kk >> 8;
  int c  = kk & 255;
  dst[i] = (__bf16)src[oc * 1024 + c * 4 + s];
}

// im2col for 2x2 stride-2 conv, fp32 x viewed as [B][4096][256], 4 channels/thread.
// col[(b*1024+p)][s*256+c] = x[b][(2*oy+ky)*64 + 2*ox+kx][c]
__global__ void k_im2col_v4(const float* __restrict__ x, __bf16* __restrict__ col, int n4) {
  int i = blockIdx.x * blockDim.x + threadIdx.x;
  if (i >= n4) return;
  int row = i >> 8;            // b*1024 + p  (256 quads per row)
  int t   = i & 255;
  int s   = t >> 6;
  int c   = (t & 63) * 4;
  int b = row >> 10, p = row & 1023;
  int ky = s >> 1, kx = s & 1;
  int oy = p >> 5, ox = p & 31;
  int nsrc = (2 * oy + ky) * 64 + (2 * ox + kx);
  float4 v = *(const float4*)&x[((size_t)b * SEQN + nsrc) * DIMC + c];
  v4bf o;
  o[0] = (__bf16)v.x; o[1] = (__bf16)v.y; o[2] = (__bf16)v.z; o[3] = (__bf16)v.w;
  *(v4bf*)&col[(size_t)row * KCONV + s * 256 + c] = o;
}

// v half of kv [2048][512] -> vT [b][h][d][m]  (bf16)
__global__ void k_transpose_v(const __bf16* __restrict__ kv, __bf16* __restrict__ vT, int n) {
  int i = blockIdx.x * blockDim.x + threadIdx.x;
  if (i >= n) return;
  int b  = i >> 18;
  int r  = i & 262143;
  int h  = r >> 15;
  int r2 = r & 32767;
  int d  = r2 >> 10;
  int m  = r2 & 1023;
  vT[i] = kv[((size_t)b * SEQM + m) * 512 + 256 + h * HDIM + d];
}

// ---------------------------------------------------------------------------
// LayerNorm over rows of 256, fp32 in -> bf16 out
// ---------------------------------------------------------------------------
__global__ void k_layernorm(const float* __restrict__ X, const float* __restrict__ g,
                            const float* __restrict__ bta, __bf16* __restrict__ out) {
  int row = blockIdx.x;
  int tid = threadIdx.x;                 // 256 threads
  float v = X[(size_t)row * DIMC + tid];
  __shared__ float s1[256], s2[256];
  s1[tid] = v; s2[tid] = v * v;
  __syncthreads();
  for (int off = 128; off > 0; off >>= 1) {
    if (tid < off) { s1[tid] += s1[tid + off]; s2[tid] += s2[tid + off]; }
    __syncthreads();
  }
  float mean = s1[0] * (1.0f / DIMC);
  float var  = s2[0] * (1.0f / DIMC) - mean * mean;
  float inv  = rsqrtf(var + 1e-5f);
  out[(size_t)row * DIMC + tid] = (__bf16)((v - mean) * inv * g[tid] + bta[tid]);
}

// ---------------------------------------------------------------------------
// bf16 WMMA GEMM: C[M x N] = A[M x K] * W[N x K]^T (+ bias)
// Block = 8 waves = 128(M) x 64(N) tile; the 64xK W slab is shared by all
// waves, double-buffered through LDS in 64x32 k-slices via async loads.
// ---------------------------------------------------------------------------
template <bool OUT_BF16>
__global__ void k_gemm_wmma(const __bf16* __restrict__ A, const __bf16* __restrict__ W,
                            void* __restrict__ Cout, const float* __restrict__ bias,
                            int M, int N, int K) {
  const int tid  = threadIdx.x;
  const int lane = tid & 31;
  const int wave = tid >> 5;
  const int cn   = lane & 15;
  const int hi   = lane >> 4;

  const int tilesN = N >> 6;
  const int tm = blockIdx.x / tilesN;
  const int tn = blockIdx.x % tilesN;
  const int row0 = tm * 128 + wave * 16;
  const int col0 = tn * 64;

  __shared__ __bf16 wtile[2][64 * 32];   // [buf][wcol][k] k-slice of W
  const int srow = tid >> 2;             // 0..63  (W row within tile)
  const int sk   = (tid & 3) * 8;        // k-chunk
  const __bf16* wsrc = W + (size_t)(col0 + srow) * K + sk;

  async_copy16(wsrc, &wtile[0][srow * 32 + sk]);
  async_wait_all();
  __syncthreads();

  v8f acc[4] = {v8f{}, v8f{}, v8f{}, v8f{}};
  const __bf16* arow = A + (size_t)(row0 + cn) * K + hi * 16;

  for (int k0 = 0; k0 < K; k0 += 32) {
    const int buf = (k0 >> 5) & 1;
    if (k0 + 32 < K)
      async_copy16(wsrc + k0 + 32, &wtile[buf ^ 1][srow * 32 + sk]);
    __builtin_prefetch((const void*)(arow + k0 + 64), 0, 1);   // global_prefetch_b8

    v16bf af = load_afrag(arow + k0);
#pragma unroll
    for (int t = 0; t < 4; ++t) {
      const __bf16* lp = &wtile[buf][(t * 16 + cn) * 32 + hi * 16];
      v16bf bfr = load_bfrag(lp);
      acc[t] = __builtin_amdgcn_wmma_f32_16x16x32_bf16(
          false, af, false, bfr, (short)0, acc[t], false, false);
    }
    async_wait_all();
    __syncthreads();
  }

#pragma unroll
  for (int t = 0; t < 4; ++t) {
    int col = col0 + t * 16 + cn;
    float bv = (bias != nullptr) ? bias[col] : 0.0f;
#pragma unroll
    for (int r = 0; r < 8; ++r) {
      int row = row0 + r + 8 * hi;       // C layout: lane half selects M 0-7 / 8-15
      float v = acc[t][r] + bv;
      if (OUT_BF16) ((__bf16*)Cout)[(size_t)row * N + col] = (__bf16)v;
      else          ((float*) Cout)[(size_t)row * N + col] = v;
    }
  }
}

// ---------------------------------------------------------------------------
// Flash attention, head_dim = 32 == WMMA K.  grid (N/128, HEADS, BATCH),
// block 256 = 8 waves, 16 queries/wave.  KV tiles of 64 staged in LDS
// (shared by all 8 waves), double-buffered with async loads.
// Softmax scale is pre-folded into q_w, so S comes out pre-scaled.
// ---------------------------------------------------------------------------
__global__ void k_attention(const __bf16* __restrict__ qflat,   // [B*4096][256] (pre-scaled)
                            const __bf16* __restrict__ kvflat,  // [B*1024][512]
                            const __bf16* __restrict__ vT,      // [B][H][32][1024]
                            __bf16* __restrict__ out) {         // [B*4096][256]
  const int tid  = threadIdx.x;
  const int lane = tid & 31;
  const int wave = tid >> 5;
  const int cn   = lane & 15;
  const int hi   = lane >> 4;
  const int h = blockIdx.y, b = blockIdx.z;
  const int q0 = blockIdx.x * 128 + wave * 16;

  __shared__ __bf16 ktile[2][64 * 32];   // [kv][d]
  __shared__ __bf16 vtile[2][32 * 64];   // [d][kv]
  __shared__ __bf16 pshare[8][16 * 64];  // per-wave P bounce (C-layout -> A-layout)
  __bf16* psh = pshare[wave];

  // Q A-fragment (16 queries x 32 dims), loaded once.
  const __bf16* qp = qflat + ((size_t)b * SEQN + q0 + cn) * DIMC + h * HDIM + hi * 16;
  v16bf qf = load_afrag(qp);

  const __bf16* kbase = kvflat + (size_t)b * SEQM * 512 + h * HDIM;
  const __bf16* vbase = vT + ((size_t)b * HEADS + h) * HDIM * SEQM;

  // Cooperative staging indices (256 threads x 16B = 4KB per tile).
  const int krow = tid >> 2;             // 0..63 kv row
  const int kch  = (tid & 3) * 8;        // d chunk
  const int vrow = tid >> 3;             // 0..31 d row
  const int vch  = (tid & 7) * 8;        // kv chunk

  async_copy16(kbase + (size_t)krow * 512 + kch, &ktile[0][krow * 32 + kch]);
  async_copy16(vbase + (size_t)vrow * SEQM + vch, &vtile[0][vrow * 64 + vch]);
  async_wait_all();
  __syncthreads();

  float mrow[8], lrow[8];
#pragma unroll
  for (int r = 0; r < 8; ++r) { mrow[r] = -3.0e38f; lrow[r] = 0.0f; }
  v8f O0 = {}, O1 = {};

  for (int kv0 = 0; kv0 < SEQM; kv0 += 64) {
    const int buf = (kv0 >> 6) & 1;
    if (kv0 + 64 < SEQM) {
      async_copy16(kbase + (size_t)(kv0 + 64 + krow) * 512 + kch,
                   &ktile[buf ^ 1][krow * 32 + kch]);
      async_copy16(vbase + (size_t)vrow * SEQM + kv0 + 64 + vch,
                   &vtile[buf ^ 1][vrow * 64 + vch]);
    }

    // S = Q K^T : 4 x (16x16), K frags straight from LDS.
    v8f S[4];
#pragma unroll
    for (int t = 0; t < 4; ++t) {
      const __bf16* lp = &ktile[buf][(t * 16 + cn) * 32 + hi * 16];
      v16bf kf = load_bfrag(lp);
      v8f z = {};
      S[t] = __builtin_amdgcn_wmma_f32_16x16x32_bf16(
          false, qf, false, kf, (short)0, z, false, false);
    }

    // Online softmax over 64 kv; row m = r + 8*hi lives in this lane-half.
#pragma unroll
    for (int r = 0; r < 8; ++r) {
      float s0 = S[0][r], s1 = S[1][r], s2 = S[2][r], s3 = S[3][r];
      float t = fmaxf(fmaxf(s0, s1), fmaxf(s2, s3));
      for (int off = 1; off < 16; off <<= 1) t = fmaxf(t, __shfl_xor(t, off, 32));
      float mnew = fmaxf(mrow[r], t);
      float corr = __expf(mrow[r] - mnew);
      float p0 = __expf(s0 - mnew), p1 = __expf(s1 - mnew);
      float p2 = __expf(s2 - mnew), p3 = __expf(s3 - mnew);
      mrow[r] = mnew;
      float ps = (p0 + p1) + (p2 + p3);
      for (int off = 1; off < 16; off <<= 1) ps += __shfl_xor(ps, off, 32);
      lrow[r] = lrow[r] * corr + ps;
      O0[r] *= corr;
      O1[r] *= corr;
      int m = r + 8 * hi;
      psh[m * 64 + cn]      = (__bf16)p0;
      psh[m * 64 + 16 + cn] = (__bf16)p1;
      psh[m * 64 + 32 + cn] = (__bf16)p2;
      psh[m * 64 + 48 + cn] = (__bf16)p3;
    }
    asm volatile("" ::: "memory");   // wave-private LDS bounce; DS ops stay in order

    // P A-frags (kv 0..31 and 32..63 of this tile).
    const __bf16* pp = psh + cn * 64 + hi * 16;
    v16bf pf0 = load_afrag(pp);
    v16bf pf1 = load_afrag(pp + 32);

    // O += P V : V frags from LDS vtile.
#pragma unroll
    for (int g = 0; g < 2; ++g) {
      v16bf pf = g ? pf1 : pf0;
      const __bf16* l0 = &vtile[buf][cn * 64 + 32 * g + hi * 16];
      const __bf16* l1 = &vtile[buf][(16 + cn) * 64 + 32 * g + hi * 16];
      O0 = __builtin_amdgcn_wmma_f32_16x16x32_bf16(
          false, pf, false, load_bfrag(l0), (short)0, O0, false, false);
      O1 = __builtin_amdgcn_wmma_f32_16x16x32_bf16(
          false, pf, false, load_bfrag(l1), (short)0, O1, false, false);
    }
    asm volatile("" ::: "memory");
    async_wait_all();
    __syncthreads();
  }

#pragma unroll
  for (int r = 0; r < 8; ++r) {
    float inv = 1.0f / lrow[r];              // uniform across the lane-half
    int nq = q0 + r + 8 * hi;
    size_t base = ((size_t)b * SEQN + nq) * DIMC + h * HDIM;
    out[base + cn]      = (__bf16)(O0[r] * inv);
    out[base + 16 + cn] = (__bf16)(O1[r] * inv);
  }
}

// ---------------------------------------------------------------------------
// Host orchestration
// ---------------------------------------------------------------------------
static inline int cdiv(int a, int b) { return (a + b - 1) / b; }

extern "C" void kernel_launch(void* const* d_in, const int* in_sizes, int n_in,
                              void* d_out, int out_size, void* d_ws, size_t ws_size,
                              hipStream_t stream) {
  const float* x      = (const float*)d_in[0];
  const float* q_w    = (const float*)d_in[3];
  const float* kv_w   = (const float*)d_in[4];
  const float* sr_w   = (const float*)d_in[5];
  const float* sr_b   = (const float*)d_in[6];
  const float* ln_g   = (const float*)d_in[7];
  const float* ln_b   = (const float*)d_in[8];
  const float* proj_w = (const float*)d_in[9];
  const float* proj_b = (const float*)d_in[10];
  float* outp = (float*)d_out;

  char* ws = (char*)d_ws;
  size_t off = 0;
  auto alloc = [&](size_t bytes) -> void* {
    off = (off + 255) & ~(size_t)255;
    void* p = ws + off;
    off += bytes;
    return p;
  };

  const int NROWS = BATCH * SEQN;   // 8192
  const int MROWS = BATCH * SEQM;   // 2048

  __bf16* xt_bf  = (__bf16*)alloc((size_t)NROWS * DIMC * 2);
  __bf16* wq_bf  = (__bf16*)alloc((size_t)DIMC * DIMC * 2);
  __bf16* wkv_bf = (__bf16*)alloc((size_t)2 * DIMC * DIMC * 2);
  __bf16* wpr_bf = (__bf16*)alloc((size_t)DIMC * DIMC * 2);
  __bf16* wsr_bf = (__bf16*)alloc((size_t)DIMC * KCONV * 2);
  __bf16* col_bf = (__bf16*)alloc((size_t)MROWS * KCONV * 2);
  float*  conv_f = (float*) alloc((size_t)MROWS * DIMC * 4);
  __bf16* xln_bf = (__bf16*)alloc((size_t)MROWS * DIMC * 2);
  __bf16* q_bf   = (__bf16*)alloc((size_t)NROWS * DIMC * 2);
  __bf16* kv_bf  = (__bf16*)alloc((size_t)MROWS * 2 * DIMC * 2);
  __bf16* vT_bf  = (__bf16*)alloc((size_t)BATCH * DIMC * SEQM * 2);
  __bf16* att_bf = (__bf16*)alloc((size_t)NROWS * DIMC * 2);
  (void)ws_size;

  const int TB = 256;

  // 1. Converts / packs (vectorized x4); softmax scale folded into q_w.
  { int n4 = NROWS * DIMC / 4;
    k_f32_to_bf16_v4<<<cdiv(n4, TB), TB, 0, stream>>>(x, xt_bf, 1.0f, n4); }
  { int n4 = DIMC * DIMC / 4;
    k_f32_to_bf16_v4<<<cdiv(n4, TB), TB, 0, stream>>>(q_w, wq_bf, SCALE_F, n4); }
  { int n4 = 2 * DIMC * DIMC / 4;
    k_f32_to_bf16_v4<<<cdiv(n4, TB), TB, 0, stream>>>(kv_w, wkv_bf, 1.0f, n4); }
  { int n4 = DIMC * DIMC / 4;
    k_f32_to_bf16_v4<<<cdiv(n4, TB), TB, 0, stream>>>(proj_w, wpr_bf, 1.0f, n4); }
  { int n = DIMC * KCONV;
    k_pack_srw<<<cdiv(n, TB), TB, 0, stream>>>(sr_w, wsr_bf, n); }
  { int n4 = MROWS * KCONV / 4;
    k_im2col_v4<<<cdiv(n4, TB), TB, 0, stream>>>(x, col_bf, n4); }

  // 2. q projection: [8192x256] = xt * (scale*q_w)^T  -> bf16
  { int blocks = (NROWS / 128) * (DIMC / 64);
    k_gemm_wmma<true><<<blocks, TB, 0, stream>>>(xt_bf, wq_bf, q_bf, nullptr,
                                                 NROWS, DIMC, DIMC); }

  // 3. conv as GEMM: [2048x256] = col * wsr^T + sr_b -> f32
  { int blocks = (MROWS / 128) * (DIMC / 64);
    k_gemm_wmma<false><<<blocks, TB, 0, stream>>>(col_bf, wsr_bf, conv_f, sr_b,
                                                  MROWS, DIMC, KCONV); }

  // 4. LayerNorm -> bf16
  k_layernorm<<<MROWS, DIMC, 0, stream>>>(conv_f, ln_g, ln_b, xln_bf);

  // 5. kv projection: [2048x512] = xln * kv_w^T -> bf16
  { int blocks = (MROWS / 128) * (2 * DIMC / 64);
    k_gemm_wmma<true><<<blocks, TB, 0, stream>>>(xln_bf, wkv_bf, kv_bf, nullptr,
                                                 MROWS, 2 * DIMC, DIMC); }

  // 6. V transpose -> [b][h][d][m]
  { int n = BATCH * DIMC * SEQM;
    k_transpose_v<<<cdiv(n, TB), TB, 0, stream>>>(kv_bf, vT_bf, n); }

  // 7. Attention
  { dim3 grid(SEQN / 128, HEADS, BATCH);
    k_attention<<<grid, TB, 0, stream>>>(q_bf, kv_bf, vT_bf, att_bf); }

  // 8. Output projection: [8192x256] = att * proj_w^T + proj_b -> f32 (d_out)
  { int blocks = (NROWS / 128) * (DIMC / 64);
    k_gemm_wmma<false><<<blocks, TB, 0, stream>>>(att_bf, wpr_bf, outp, proj_b,
                                                  NROWS, DIMC, DIMC); }
  (void)in_sizes; (void)n_in; (void)out_size;
}
